// Q4Linear_21242908246547
// MI455X (gfx1250) — matrix operand order repaired and checked by reading, not compiled, not run on hardware
//
#include <hip/hip_runtime.h>

// ---------------------------------------------------------------------------
// W4A16 quantized linear for MI455X (gfx1250, wave32, WMMA).
//   out[M,N] = x[M,K] * W^T + bias,  W dequantized per-128-group: w = q*s + z
// M=8192, K=4096, N=11008.  739 GFLOP vs ~0.5 GB HBM traffic -> compute bound,
// so: dequant-to-f16-in-LDS + v_wmma_f32_16x16x32_f16, double-buffered LDS,
// x tile staged with GLOBAL_LOAD_ASYNC_TO_LDS_B128 (ASYNCcnt) overlapped with
// the WMMA stream of the previous chunk.
// ---------------------------------------------------------------------------

typedef __attribute__((ext_vector_type(16))) _Float16 v16h;
typedef __attribute__((ext_vector_type(8)))  _Float16 v8h;
typedef __attribute__((ext_vector_type(2)))  _Float16 v2h;
typedef __attribute__((ext_vector_type(8)))  float    v8f;

#define IN_F   4096
#define OUT_F  11008
#define BM     128
#define BN     128
#define BK     128
#define NCHUNK (IN_F / BK)   // 32
#define XTILE  (BM * BK)     // halves per x buffer
#define WTILE  (BN * BK)     // halves per w buffer

// Async copy 16B/lane from global to LDS. NOTE: the instruction's immediate
// offset is added to BOTH the LDS address (VDST) and the memory address, so a
// fixed (ldsa, voff) pair plus offsets 0..112 streams 128B/lane.
#define ASYNC_CP16(ldsa, voff, basep, OFF)                                    \
  asm volatile("global_load_async_to_lds_b128 %0, %1, %2 offset:" OFF         \
               :: "v"(ldsa), "v"(voff), "s"(basep) : "memory")

static __device__ __forceinline__ void async_x_tile(unsigned ldsa, unsigned voff,
                                                    const _Float16* base) {
  ASYNC_CP16(ldsa, voff, base, "0");
  ASYNC_CP16(ldsa, voff, base, "16");
  ASYNC_CP16(ldsa, voff, base, "32");
  ASYNC_CP16(ldsa, voff, base, "48");
  ASYNC_CP16(ldsa, voff, base, "64");
  ASYNC_CP16(ldsa, voff, base, "80");
  ASYNC_CP16(ldsa, voff, base, "96");
  ASYNC_CP16(ldsa, voff, base, "112");
}

static __device__ __forceinline__ void wait_async0() {
  asm volatile("s_wait_asynccnt 0x0" ::: "memory");
}

// Two 4-bit weights (low nibble first) -> packed f16 pair.
// nibble|0x6400 is exactly (1024+n) in f16; (h - 1032) is exact (= n-8);
// the packed f16 FMA then rounds once: matches the f32 reference to <=1 ulp
// double-rounding.  Lowers to v_pk_add_f16 + v_pk_fma_f16.
static __device__ __forceinline__ unsigned deq2(int b, v2h sv, v2h zv) {
  unsigned u = ((unsigned)b & 15u) | (((unsigned)b & 0xF0u) << 12) | 0x64006400u;
  v2h hv = __builtin_bit_cast(v2h, u);
  const v2h c1032 = {(_Float16)1032.0f, (_Float16)1032.0f};
  v2h w = (hv - c1032) * sv + zv;
  return __builtin_bit_cast(unsigned, w);
}

__global__ __launch_bounds__(256)
void q4linear_wmma_kernel(const _Float16* __restrict__ x,
                          const int*      __restrict__ qweight,
                          const _Float16* __restrict__ scales,
                          const _Float16* __restrict__ zeros,
                          const _Float16* __restrict__ bias,
                          float*          __restrict__ out) {
  // Dynamic LDS: 2 x 32KB x-buffers + 2 x 32KB w-buffers = 128 KB.
  extern __shared__ _Float16 smem[];
  _Float16* lds_x = smem;                 // [2][BM][BK]
  _Float16* lds_w = smem + 2 * XTILE;     // [2][BN][BK]

  const int t  = threadIdx.x;             // 0..255 (8 wave32s)
  const int m0 = blockIdx.y * BM;
  const int n0 = blockIdx.x * BN;

  // ---- loader/dequant role: thread t owns half a row of the tile ----
  const int row   = t >> 1;               // 0..127
  const int kh    = (t & 1) * 64;         // 0 or 64 halves within chunk
  const int n_row = n0 + row;

  // per-thread staging addresses
  const unsigned xvoff0 = (unsigned)(((m0 + row) * IN_F + kh) * 2);  // bytes
  const unsigned ldsx_addr[2] = {
      (unsigned)(size_t)(lds_x + 0 * XTILE + row * BK + kh),
      (unsigned)(size_t)(lds_x + 1 * XTILE + row * BK + kh)};
  // 32 packed bytes (as int32s) per thread per chunk; chunk stride = 16 int4
  const int4* qrow = (const int4*)(qweight + (((size_t)n_row * IN_F + kh) >> 1));

  // ---- compute role: wave grid 4 (M) x 2 (N); each wave 32x64 ----
  const int wv   = t >> 5;
  const int lane = t & 31;
  const int l16  = lane & 15;
  const int hl   = lane >> 4;             // 0: lanes 0-15, 1: lanes 16-31
  const int wm   = (wv & 3) * 32;
  const int wn   = (wv >> 2) * 64;

  v8f acc[2][4];
  {
    v8f vzero = {};
    #pragma unroll
    for (int im = 0; im < 2; ++im)
      #pragma unroll
      for (int in = 0; in < 4; ++in)
        acc[im][in] = vzero;
  }

  // ---------------- prologue: stage chunk 0 into buffer 0 ----------------
  {
    async_x_tile(ldsx_addr[0], xvoff0, x);
    int4 qr[8];
    #pragma unroll
    for (int i = 0; i < 8; ++i) qr[i] = qrow[i];
    const int g = n_row * (IN_F / 128) + 0;
    const _Float16 s = scales[g], z = zeros[g];
    const v2h sv = {s, s}, zv = {z, z};
    int4* dst = (int4*)(lds_w + 0 * WTILE + row * BK + kh);
    #pragma unroll
    for (int i = 0; i < 8; ++i) {
      int4 q = qr[i], o;
      o.x = (int)deq2(q.x, sv, zv);
      o.y = (int)deq2(q.y, sv, zv);
      o.z = (int)deq2(q.z, sv, zv);
      o.w = (int)deq2(q.w, sv, zv);
      dst[i] = o;
    }
    wait_async0();
    __syncthreads();
  }

  // ---------------- main pipelined loop ----------------
  for (int kc = 0; kc < NCHUNK; ++kc) {
    const int cur = kc & 1;
    const int nxt = cur ^ 1;
    const bool has_next = (kc + 1 < NCHUNK);

    // Issue next chunk's staging traffic BEFORE this chunk's WMMAs so the
    // async copies and qweight loads overlap the matrix-core stream.
    int4 qr[8];
    if (has_next) {
      async_x_tile(ldsx_addr[nxt], xvoff0 + (unsigned)(kc + 1) * (BK * 2), x);
      #pragma unroll
      for (int i = 0; i < 8; ++i) qr[i] = qrow[(kc + 1) * 16 + i];
    }

    // -------- compute chunk kc from buffer cur --------
    const _Float16* xb = lds_x + cur * XTILE;
    const _Float16* wb = lds_w + cur * WTILE;
    #pragma unroll
    for (int kk = 0; kk < 4; ++kk) {
      const int kb = kk * 32;

      // A frag (16x32 f16): lane hl=0 holds K=[0..15];
      // hl=1 holds K=[16..31] -> one 32B contiguous LDS read.
      v16h a[2];
      #pragma unroll
      for (int im = 0; im < 2; ++im) {
        const _Float16* base = xb + (wm + im * 16 + l16) * BK + kb + hl * 16;
        a[im] = *(const v16h*)base;
      }
      // B frag (32x16 f16): lane holds contiguous B[k..k+15][n] ==
      // lds_w[n][k..k+15] -> one 32B LDS read.
      v16h b[4];
      #pragma unroll
      for (int in = 0; in < 4; ++in) {
        const _Float16* base = wb + (wn + in * 16 + l16) * BK + kb + hl * 16;
        b[in] = *(const v16h*)base;
      }

      #pragma unroll
      for (int im = 0; im < 2; ++im)
        #pragma unroll
        for (int in = 0; in < 4; ++in)
          acc[im][in] = __builtin_amdgcn_wmma_f32_16x16x32_f16(
              false, a[im], false, b[in], (short)0, acc[im][in], false, false);
    }

    // -------- finish staging chunk kc+1 into buffer nxt --------
    if (has_next) {
      const int g = n_row * (IN_F / 128) + (kc + 1);
      const _Float16 s = scales[g], z = zeros[g];
      const v2h sv = {s, s}, zv = {z, z};
      int4* dst = (int4*)(lds_w + nxt * WTILE + row * BK + kh);
      #pragma unroll
      for (int i = 0; i < 8; ++i) {
        int4 q = qr[i], o;
        o.x = (int)deq2(q.x, sv, zv);
        o.y = (int)deq2(q.y, sv, zv);
        o.z = (int)deq2(q.z, sv, zv);
        o.w = (int)deq2(q.w, sv, zv);
        dst[i] = o;
      }
      wait_async0();     // our asm async ops are invisible to the compiler
      __syncthreads();
    }
  }

  // ---------------- epilogue: bias add + fp32 store ----------------
  // D layout (16x16 f32): VGPR r, lanes 0-15 -> M=r, lanes 16-31 -> M=8+r.
  #pragma unroll
  for (int in = 0; in < 4; ++in) {
    const int n = n0 + wn + in * 16 + l16;
    const float bv = (float)bias[n];
    #pragma unroll
    for (int im = 0; im < 2; ++im) {
      const int mb = m0 + wm + im * 16 + hl * 8;
      #pragma unroll
      for (int r = 0; r < 8; ++r) {
        out[(size_t)(mb + r) * OUT_F + n] = acc[im][in][r] + bv;
      }
    }
  }
}

extern "C" void kernel_launch(void* const* d_in, const int* in_sizes, int n_in,
                              void* d_out, int out_size, void* d_ws, size_t ws_size,
                              hipStream_t stream) {
  const _Float16* x  = (const _Float16*)d_in[0];
  const int*      qw = (const int*)d_in[1];
  const _Float16* sc = (const _Float16*)d_in[2];
  const _Float16* zr = (const _Float16*)d_in[3];
  const _Float16* bs = (const _Float16*)d_in[4];
  float* out = (float*)d_out;

  const int M = in_sizes[0] / IN_F;                  // 8192
  dim3 grid(OUT_F / BN, M / BM);                     // (86, 64)
  const size_t lds_bytes = 2 * (XTILE + WTILE) * sizeof(_Float16);  // 128 KB
  q4linear_wmma_kernel<<<grid, 256, lds_bytes, stream>>>(x, qw, sc, zr, bs, out);
}
